// GATConv_90529320665790
// MI455X (gfx1250) — matrix operand order repaired and checked by reading, not compiled, not run on hardware
//
#include <hip/hip_runtime.h>
#include <hip/hip_bf16.h>
#include <stdint.h>

#define N     8192
#define NEMB  256
#define NHID  64
#define CHUNK 256
#define NCH   (N / CHUNK)
#define NEG_INF_F (-9e15f)

typedef __attribute__((ext_vector_type(8)))  float    v8f;
typedef __attribute__((ext_vector_type(2)))  float    v2f;
typedef __attribute__((ext_vector_type(16))) __bf16   v16bf;
typedef __attribute__((ext_vector_type(4)))  unsigned u32x4;
typedef __attribute__((ext_vector_type(8)))  int      i32x8;
typedef __attribute__((ext_vector_type(4)))  int      i32x4;

union Frag16 { uint4 q[2]; v16bf v; };
union Pack4  { __bf16 h[4]; uint2 u; };

#if __has_builtin(__builtin_amdgcn_tensor_load_to_lds) && \
    __has_builtin(__builtin_amdgcn_s_wait_tensorcnt)
#define USE_TDM 1
#else
#define USE_TDM 0
#endif

// ---------------------------------------------------------------------------
// K1: h = x @ Ww + Wb   (f32 WMMA 16x16x4), also writes hbT = h^T in bf16.
// ---------------------------------------------------------------------------
__global__ __launch_bounds__(128) void k_hproj(const float* __restrict__ x,
                                               const float* __restrict__ Ww,
                                               const float* __restrict__ Wb,
                                               float* __restrict__ h,
                                               __bf16* __restrict__ hbT) {
  const int lane  = threadIdx.x & 31;
  const int wave  = threadIdx.x >> 5;     // col tile 0..3
  const int R     = blockIdx.x * 16;
  const int mrow  = lane & 15;
  const int khalf = lane >> 4;
  const int n     = wave * 16 + mrow;
#if __has_builtin(__builtin_amdgcn_wmma_f32_16x16x4_f32)
  v8f acc = {};
  for (int k = 0; k < NEMB; k += 4) {
    const float* ap = &x[(size_t)(R + mrow) * NEMB + k + 2 * khalf];
    v2f a; a.x = ap[0]; a.y = ap[1];
    v2f b;
    b.x = Ww[(size_t)(k + 2 * khalf) * NHID + n];
    b.y = Ww[(size_t)(k + 2 * khalf + 1) * NHID + n];
    acc = __builtin_amdgcn_wmma_f32_16x16x4_f32(false, a, false, b,
                                                (short)0, acc, false, false);
  }
  const float bias = Wb[n];
  const int mh = khalf * 8;
#pragma unroll
  for (int r = 0; r < 8; ++r) {
    const int row = R + mh + r;
    const float v = acc[r] + bias;
    h[(size_t)row * NHID + n] = v;
    hbT[(size_t)n * N + row]  = (__bf16)v;
  }
#else
  for (int e = threadIdx.x; e < 16 * NHID; e += 128) {
    const int rr = e >> 6, nn = e & 63;
    float s = Wb[nn];
    for (int k = 0; k < NEMB; ++k)
      s += x[(size_t)(R + rr) * NEMB + k] * Ww[(size_t)k * NHID + nn];
    h[(size_t)(R + rr) * NHID + nn] = s;
    hbT[(size_t)nn * N + (R + rr)]  = (__bf16)s;
  }
#endif
}

// ---------------------------------------------------------------------------
// K2: f_src[i] = h[i].a1 + ab ; f_dst[i] = h[i].a2
// ---------------------------------------------------------------------------
__global__ __launch_bounds__(256) void k_attvec(const float* __restrict__ h,
                                                const float* __restrict__ a1,
                                                const float* __restrict__ a2,
                                                const float* __restrict__ ab,
                                                float* __restrict__ fsrc,
                                                float* __restrict__ fdst) {
  const int i = blockIdx.x * 256 + threadIdx.x;
  if (i >= N) return;
  float s1 = ab[0], s2 = 0.f;
  const float* hp = &h[(size_t)i * NHID];
#pragma unroll 8
  for (int j = 0; j < NHID; ++j) {
    const float hv = hp[j];
    s1 += hv * a1[j];
    s2 += hv * a2[j];
  }
  fsrc[i] = s1;
  fdst[i] = s2;
}

// ---------------------------------------------------------------------------
// K3: rowcnt[i] = popcount of adj row i   (streams 256 MB once)
// ---------------------------------------------------------------------------
__global__ __launch_bounds__(256) void k_rowcnt(const int* __restrict__ adj,
                                                unsigned* __restrict__ rowcnt) {
  const int row = blockIdx.x;
  const int4* ap = (const int4*)&adj[(size_t)row * N];
  int c = 0;
  for (int j = threadIdx.x; j < N / 4; j += 256) {
    const int4 v = ap[j];
    c += (v.x > 0) + (v.y > 0) + (v.z > 0) + (v.w > 0);
  }
  __shared__ int red[256];
  red[threadIdx.x] = c;
  __syncthreads();
  for (int s = 128; s; s >>= 1) {
    if (threadIdx.x < (unsigned)s) red[threadIdx.x] += red[threadIdx.x + s];
    __syncthreads();
  }
  if (threadIdx.x == 0) rowcnt[row] = (unsigned)red[0];
}

// ---------------------------------------------------------------------------
// K4: exclusive scan of rowcnt -> rowoff (global True-rank base per row)
// ---------------------------------------------------------------------------
__global__ __launch_bounds__(256) void k_scan(const unsigned* __restrict__ rowcnt,
                                              unsigned* __restrict__ rowoff) {
  __shared__ unsigned ts[256];
  const int t = threadIdx.x, base = t * 32;
  unsigned s = 0;
  for (int e = 0; e < 32; ++e) s += rowcnt[base + e];
  ts[t] = s;
  __syncthreads();
  if (t == 0) {
    unsigned run = 0;
    for (int i = 0; i < 256; ++i) { const unsigned v = ts[i]; ts[i] = run; run += v; }
  }
  __syncthreads();
  unsigned run = ts[t];
  for (int e = 0; e < 32; ++e) { rowoff[base + e] = run; run += rowcnt[base + e]; }
}

// ---------------------------------------------------------------------------
// TDM: DMA a 16 x CHUNK int32 tile (row stride N) from global into LDS.
// D# per CDNA5 ISA ch.8: group0 = {count|flags, lds_addr, global_addr, type=2},
// group1 = {data_size=4B, tensor 8192x8192, tile CHUNKx16, stride0=8192}.
// Groups 2/3 zero (2D tile, tile_dim2=0).
// ---------------------------------------------------------------------------
#if USE_TDM
__device__ __forceinline__ void tdm_load_adj(unsigned ldsByteOff,
                                             const int* __restrict__ gsrc) {
  const unsigned long long ga = (unsigned long long)(uintptr_t)gsrc;
  u32x4 g0;
  g0.x = 1u;                                                   // count=1 (valid)
  g0.y = ldsByteOff;                                           // lds_addr
  g0.z = (unsigned)ga;                                         // global_addr lo
  g0.w = (unsigned)((ga >> 32) & 0x01FFFFFFull) | (2u << 30);  // addr hi | type=2
  i32x8 g1;
  g1[0] = (int)(2u << 16);                            // data_size = 4 bytes
  g1[1] = (int)((unsigned)(N & 0xFFFF) << 16);        // tensor_dim0[15:0] << 16
  g1[2] = (int)((N >> 16) | ((unsigned)(N & 0xFFFF) << 16)); // dim0 hi | dim1 lo
  g1[3] = (int)((N >> 16) | ((unsigned)CHUNK << 16)); // dim1 hi | tile_dim0
  g1[4] = 16;                                         // tile_dim1=16, tile_dim2=0
  g1[5] = N;                                          // tensor_dim0_stride lo
  g1[6] = 0;
  g1[7] = 0;
  const i32x4 z4 = {0, 0, 0, 0};
#if __has_include(<hip/amd_detail/amd_gfx1250_TDM.h>)
  const i32x8 z8 = {0, 0, 0, 0, 0, 0, 0, 0};          // therock: 6-arg builtin
  __builtin_amdgcn_tensor_load_to_lds(g0, g1, z4, z4, z8, 0);
#else
  __builtin_amdgcn_tensor_load_to_lds(g0, g1, z4, z4, 0);      // ROCm 7.2: 5-arg
#endif
}
#endif

// ---------------------------------------------------------------------------
// K5: fused masked-scatter scores + online (flash) softmax + att@h via
// v_wmma_f32_16x16x32_bf16 + ELU.  Block = 512 threads = 16 waves, 16 rows.
// adj tiles stream into LDS via the Tensor Data Mover, double-buffered on
// TENSORcnt, overlapping the HBM stream with the WMMA phase.
// ---------------------------------------------------------------------------
__global__ __launch_bounds__(512) void k_gat(const int* __restrict__ adj,
                                             const float* __restrict__ fsrc,
                                             const float* __restrict__ fdst,
                                             const unsigned* __restrict__ rowoff,
                                             const __bf16* __restrict__ hbT,
                                             float* __restrict__ out) {
  __shared__ __align__(16) int    s_adj[2][16][CHUNK];   // 32 KB (double buffer)
  __shared__ __align__(16) __bf16 s_sc[16][CHUNK];       //  8 KB probs tile
  __shared__ __align__(16) float  s_red[16 * 32 * 8];    // 16 KB k-part reduce
  __shared__ float s_alpha[16];
  __shared__ float s_invl[16];

  const int lane  = threadIdx.x & 31;
  const int wave  = threadIdx.x >> 5;   // 0..15
  const int row0  = blockIdx.x * 16;
  const int myrow = row0 + wave;
  const int ct    = wave & 3;           // GEMM column tile
  const int kp    = wave >> 2;          // GEMM k partition
  const int mrow  = lane & 15;
  const int khalf = lane >> 4;
  const int mh    = khalf * 8;
  const int nOut  = ct * 16 + mrow;

  const int* adjRow  = &adj[(size_t)myrow * N];
  const int* adjBase = &adj[(size_t)row0 * N];   // 16-row block origin
  (void)adjRow; (void)adjBase;
  const unsigned offBase = rowoff[myrow];

  unsigned runCnt = 0;
  float rowMax = NEG_INF_F;
  float rowL   = 0.f;
  v8f acc = {};

#if USE_TDM
  if (wave == 0)
    tdm_load_adj((unsigned)(uintptr_t)&s_adj[0][0][0], adjBase);
#endif

  for (int c = 0; c < NCH; ++c) {
    const int col0 = c * CHUNK;
    const int buf  = c & 1;
    (void)buf;

#if USE_TDM
    // Prefetch chunk c+1 via TDM while this chunk computes; wait for chunk c.
    if (wave == 0) {
      if (c + 1 < NCH) {
        tdm_load_adj((unsigned)(uintptr_t)&s_adj[(c + 1) & 1][0][0],
                     adjBase + (size_t)(c + 1) * CHUNK);
        __builtin_amdgcn_s_wait_tensorcnt(1);   // load(c) done, load(c+1) flying
      } else {
        __builtin_amdgcn_s_wait_tensorcnt(0);
      }
    }
    __syncthreads();   // adj buffer ready; also fences prior GEMM's s_sc reads
#else
    __syncthreads();   // fences prior GEMM's s_sc reads before overwrite
#endif

    // ---- score pass: 4 columns/lane/iter, ballot-ranked masked_scatter
    float chMax = NEG_INF_F;
    unsigned cnt = runCnt;
#pragma unroll
    for (int it = 0; it < CHUNK / 128; ++it) {
      const int cb = it * 128 + lane * 4;
#if USE_TDM
      const int4 av = *(const int4*)&s_adj[buf][wave][cb];
#else
      const int4 av = *(const int4*)&adjRow[col0 + cb];
#if defined(__gfx1250__)
      if (col0 + cb + CHUNK < N) __builtin_prefetch(&adjRow[col0 + cb + CHUNK], 0, 0);
#endif
#endif
      const bool m0 = av.x > 0, m1 = av.y > 0, m2 = av.z > 0, m3 = av.w > 0;
      const unsigned b0 = (unsigned)__ballot(m0);
      const unsigned b1 = (unsigned)__ballot(m1);
      const unsigned b2 = (unsigned)__ballot(m2);
      const unsigned b3 = (unsigned)__ballot(m3);
      const unsigned below = (1u << lane) - 1u;
      unsigned r = cnt + __popc(b0 & below) + __popc(b1 & below)
                       + __popc(b2 & below) + __popc(b3 & below);
      Pack4 pk;
#pragma unroll
      for (int j = 0; j < 4; ++j) {
        const bool m = (j == 0) ? m0 : (j == 1) ? m1 : (j == 2) ? m2 : m3;
        float s = NEG_INF_F;
        if (m) {
          const unsigned g = offBase + r;
          s = fsrc[g >> 13] + fdst[g & (N - 1)];   // e.flatten()[g] (rank-1 e)
          s = s > 0.f ? s : 0.01f * s;             // leaky_relu
          ++r;
        }
        const __bf16 sb = (__bf16)s;               // round BEFORE max
        pk.h[j] = sb;
        chMax = fmaxf(chMax, (float)sb);
      }
      *(uint2*)&s_sc[wave][cb] = pk.u;
      cnt += __popc(b0) + __popc(b1) + __popc(b2) + __popc(b3);
    }
    runCnt = cnt;
#pragma unroll
    for (int o = 16; o; o >>= 1) chMax = fmaxf(chMax, __shfl_xor(chMax, o, 32));
    const float newMax = fmaxf(rowMax, chMax);

    // ---- exp pass: overwrite scores with un-normalized probs (bf16)
    float sum = 0.f;
#pragma unroll
    for (int it = 0; it < CHUNK / 128; ++it) {
      const int cb = it * 128 + lane * 4;
      Pack4 pk;
      pk.u = *(const uint2*)&s_sc[wave][cb];
#pragma unroll
      for (int j = 0; j < 4; ++j) {
        const __bf16 eb = (__bf16)__expf((float)pk.h[j] - newMax);
        pk.h[j] = eb;
        sum += (float)eb;
      }
      *(uint2*)&s_sc[wave][cb] = pk.u;
    }
#pragma unroll
    for (int o = 16; o; o >>= 1) sum += __shfl_xor(sum, o, 32);
    const float alpha = __expf(rowMax - newMax);
    rowL = rowL * alpha + sum;
    rowMax = newMax;
    if (lane == 0) s_alpha[wave] = alpha;
    __syncthreads();

    // ---- GEMM phase: rescale accumulators (flash-style), then WMMA
#pragma unroll
    for (int r = 0; r < 8; ++r) acc[r] *= s_alpha[mh + r];

#pragma unroll
    for (int ks = 0; ks < (CHUNK / 4) / 32; ++ks) {
      const int kloc = kp * (CHUNK / 4) + ks * 32;
      Frag16 af, bfm;
      // A (16x32 bf16): lane M=mrow; K runs [khalf*8,+8) and [16+khalf*8,+8)
      af.q[0] = *(const uint4*)&s_sc[mrow][kloc + khalf * 8];
      af.q[1] = *(const uint4*)&s_sc[mrow][kloc + 16 + khalf * 8];
      // B (32x16 bf16): lane N=nOut; K runs [khalf*8,+8) and [16+khalf*8,+8) in hbT
      const int kglob = col0 + kloc + khalf * 8;
      const uint4* bp = (const uint4*)&hbT[(size_t)nOut * N + kglob];
      bfm.q[0] = bp[0];
      bfm.q[1] = bp[2];
      acc = __builtin_amdgcn_wmma_f32_16x16x32_bf16(false, af.v, false, bfm.v,
                                                    (short)0, acc, false, false);
    }
    // no trailing barrier: next chunk's top barrier fences s_sc reuse
  }

  if (lane == 0) s_invl[wave] = 1.0f / rowL;
  // ---- cross-k-partition reduce + softmax normalization + ELU epilogue
  {
    float* p = &s_red[((size_t)wave * 32 + lane) * 8];
#pragma unroll
    for (int r = 0; r < 8; ++r) p[r] = acc[r];
  }
  __syncthreads();
  if (wave < 4) {                      // wave == its col tile (kp == 0)
#pragma unroll
    for (int r = 0; r < 8; ++r) {
      float v = s_red[((wave     ) * 32 + lane) * 8 + r]
              + s_red[((wave +  4) * 32 + lane) * 8 + r]
              + s_red[((wave +  8) * 32 + lane) * 8 + r]
              + s_red[((wave + 12) * 32 + lane) * 8 + r];
      const int M = mh + r;
      v *= s_invl[M];
      v = v > 0.f ? v : (__expf(v) - 1.f);
      out[(size_t)(row0 + M) * NHID + wave * 16 + mrow] = v;
    }
  }
}

// ---------------------------------------------------------------------------
extern "C" void kernel_launch(void* const* d_in, const int* in_sizes, int n_in,
                              void* d_out, int out_size, void* d_ws, size_t ws_size,
                              hipStream_t stream) {
  const float* x   = (const float*)d_in[0];
  const int*   adj = (const int*)  d_in[1];
  const float* Ww  = (const float*)d_in[2];
  const float* Wb  = (const float*)d_in[3];
  const float* a1  = (const float*)d_in[4];
  const float* a2  = (const float*)d_in[5];
  const float* ab  = (const float*)d_in[6];
  float* out = (float*)d_out;

  char* ws = (char*)d_ws;
  float*    h      = (float*)ws;                                   // 2 MB
  __bf16*   hbT    = (__bf16*)(ws + (size_t)2 * 1024 * 1024);      // 1 MB
  float*    fsrc   = (float*)(ws + (size_t)3 * 1024 * 1024);       // 32 KB
  float*    fdst   = (float*)(ws + (size_t)3 * 1024 * 1024 + 32 * 1024);
  unsigned* rowcnt = (unsigned*)(ws + (size_t)3 * 1024 * 1024 + 64 * 1024);
  unsigned* rowoff = (unsigned*)(ws + (size_t)3 * 1024 * 1024 + 96 * 1024);

  k_hproj <<<N / 16, 128, 0, stream>>>(x, Ww, Wb, h, hbT);
  k_attvec<<<N / 256, 256, 0, stream>>>(h, a1, a2, ab, fsrc, fdst);
  k_rowcnt<<<N, 256, 0, stream>>>(adj, rowcnt);
  k_scan  <<<1, 256, 0, stream>>>(rowcnt, rowoff);
  k_gat   <<<N / 16, 512, 0, stream>>>(adj, fsrc, fdst, rowoff, hbT, out);
}
